// NeuralControlCritic_47407849013835
// MI455X (gfx1250) — matrix operand order repaired and checked by reading, not compiled, not run on hardware
//
#include <hip/hip_runtime.h>
#include <hip/hip_bf16.h>
#include <math.h>

// ---------------------------------------------------------------------------
// NeuralControlCritic (Mamba-2 stack) for MI455X / gfx1250, wave32.
//   B=4, S=512 -> M = 2048 token rows
//   D_MODEL=1024, D_INNER=2048, D_STATE=128, H=32, P=64, DCONV=4
//   D_IN_PROJ = 4384 (z:2048 | xBC:2304 | dt:32), CONV_DIM = 2304
// All GEMM operands are bf16 in global memory (weights converted once per
// launch, activations produced in bf16); GEMMs run v_wmma_f32_16x16x32_bf16
// with the A tile staged by GLOBAL_LOAD_ASYNC_TO_LDS_B128 (ASYNCcnt).
// The sequential selective scan keeps fp32 state in registers and hides its
// per-step B/C staging behind compute with double-buffered async-to-LDS.
// ---------------------------------------------------------------------------

#define MROWS   2048
#define DMODEL  1024
#define DINNER  2048
#define DSTATE  128
#define NHEADS  32
#define HEADDIM 64
#define DINPROJ 4384
#define CONVDIM 2304
#define SEQLEN  512

typedef float  v8f   __attribute__((ext_vector_type(8)));
typedef __bf16 v16bf __attribute__((ext_vector_type(16)));
typedef unsigned int u32x4 __attribute__((ext_vector_type(4)));

union FragU { u32x4 q[2]; v16bf v; };

__device__ __forceinline__ unsigned short f2bf(float f) {
    unsigned u = __float_as_uint(f);
    u += 0x7FFFu + ((u >> 16) & 1u);          // round-to-nearest-even
    return (unsigned short)(u >> 16);
}
__device__ __forceinline__ unsigned pack2(float a, float b) {
    return (unsigned)f2bf(a) | ((unsigned)f2bf(b) << 16);
}
__device__ __forceinline__ float bf2f(unsigned short u) {
    return __uint_as_float((unsigned)u << 16);
}
__device__ __forceinline__ float siluf(float x) {
    return x / (1.0f + __expf(-x));
}

// ASYNCcnt-tracked DMA: global -> LDS (ISA 15.18.3 ops 96/98).
__device__ __forceinline__ void async_ld_b128(unsigned lds_off, const void* gptr) {
    asm volatile("global_load_async_to_lds_b128 %0, %1, off"
                 :: "v"(lds_off), "v"((unsigned long long)(uintptr_t)gptr)
                 : "memory");
}
__device__ __forceinline__ void async_ld_b32(unsigned lds_off, const void* gptr) {
    asm volatile("global_load_async_to_lds_b32 %0, %1, off"
                 :: "v"(lds_off), "v"((unsigned long long)(uintptr_t)gptr)
                 : "memory");
}
__device__ __forceinline__ void wait_async0() {
    asm volatile("s_wait_asynccnt 0x0" ::: "memory");
}

// ---------------------------------------------------------------------------
// fp32 -> bf16 bulk conversion (8 elements / thread, b128 in, b128 out)
// ---------------------------------------------------------------------------
__global__ void k_f2bf(const float* __restrict__ src,
                       unsigned short* __restrict__ dst, int n8)
{
    int i = blockIdx.x * 256 + threadIdx.x;
    if (i >= n8) return;
    const float4* s = (const float4*)src + (size_t)i * 2;
    float4 a = s[0], b = s[1];
    u32x4 o;
    o.x = pack2(a.x, a.y); o.y = pack2(a.z, a.w);
    o.z = pack2(b.x, b.y); o.w = pack2(b.z, b.w);
    *((u32x4*)dst + i) = o;
}

// ---------------------------------------------------------------------------
// Tiled bf16 WMMA GEMM:  C = act(A[M,K] @ W[K,N] + bias)
//   A,W bf16 row-major; C written fp32 and/or bf16 (nullable pointers).
//   block tile 128x128, K step 32; 256 threads = 8 waves, wave tile 32x64.
//   A tile staged via async DMA to LDS; W tile transpose-staged via VGPRs
//   with v_perm_b32 pair packing.
//   REQUIREMENT: W must point into a buffer with >= 4 KB of tail padding
//   (we always pass the WB scratch): the N-tail tile reads past row ends and
//   the garbage only feeds accumulators whose stores are col-guarded.
//   M mult of 128, K mult of 32 (all call sites).
// ---------------------------------------------------------------------------
#define TM 128
#define TN 128
#define TK 32
#define LDP (TK + 8)   // padded LDS row (shorts): 80 B, keeps 16-B chunks aligned

__global__ __launch_bounds__(256)
void k_gemm_bf16(const unsigned short* __restrict__ A,
                 const unsigned short* __restrict__ W,
                 const float* __restrict__ bias,
                 float* __restrict__ Cf, unsigned short* __restrict__ Cb,
                 int M, int N, int K, int relu)
{
    __shared__ unsigned short As[TM][LDP];   // A tile, row-major [m][k]
    __shared__ unsigned short Bs[TN][LDP];   // W tile transposed [n][k]

    const int t    = threadIdx.x;
    const int lane = t & 31;
    const int wave = t >> 5;
    const int row0 = blockIdx.y * TM;
    const int col0 = blockIdx.x * TN;
    const int wrow = (wave >> 1) * 32;       // 4 waves down
    const int wcol = (wave & 1)  * 64;       // 2 waves across

    const int hi = lane >> 4;                // K-half select (ISA 7.12.2)
    const int lm = lane & 15;                // M (A) / N (B) index

    v8f acc[2][4];
    #pragma unroll
    for (int i = 0; i < 2; ++i)
        #pragma unroll
        for (int j = 0; j < 4; ++j) { v8f z = {}; acc[i][j] = z; }

    // A async mapping: chunk c = 2t,2t+1; 16-B chunks, 4 per 32-elem k-row
    const int arow = t >> 1;                 // 0..127
    const int aseg = (t & 1) * 2;            // 0 | 2  (16-B chunk index)
    const unsigned lds_a0 =
        (unsigned)(unsigned long long)(void*)&As[arow][aseg * 8];

    // W mapping: 2 k-rows x 8 n per thread
    const int k2 = (t & 15) * 2;             // even k row in tile
    const int n0 = (t >> 4) * 8;             // n base (0..120)
    const unsigned short* ap = A + (size_t)(row0 + arow) * K + aseg * 8;
    const unsigned short* wp = W + (size_t)k2 * N + col0 + n0;

    for (int k0 = 0; k0 < K; k0 += TK) {
        // ---- A tile: async DMA straight into LDS (2 x b128 per thread)
        async_ld_b128(lds_a0,      ap);
        async_ld_b128(lds_a0 + 16, ap + 8);

        // ---- W tile: 2 k-rows x 8 n (b128 each), v_perm pair-pack transpose
        u32x4 w0 = *(const u32x4*)wp;
        u32x4 w1 = *(const u32x4*)(wp + N);
        #pragma unroll
        for (int jw = 0; jw < 4; ++jw) {
            unsigned lo = w0[jw], hiw = w1[jw];
            // dst = { hi.b1, hi.b0, lo.b1, lo.b0 } etc: one v_perm_b32 each
            *(unsigned*)&Bs[n0 + 2 * jw][k2] =
                __builtin_amdgcn_perm(hiw, lo, 0x05040100u);
            *(unsigned*)&Bs[n0 + 2 * jw + 1][k2] =
                __builtin_amdgcn_perm(hiw, lo, 0x07060302u);
        }

        if (k0 + TK < K) {                   // global_prefetch_b8 next tiles
            __builtin_prefetch(ap + TK, 0, 1);
            __builtin_prefetch(wp + TK * N, 0, 1);
        }
        ap += TK;
        wp += (size_t)TK * N;

        wait_async0();
        __syncthreads();

        // ---- fragments (two 16-B LDS chunks each) and 8 WMMAs
        FragU afr[2], bfr[4];
        #pragma unroll
        for (int mt = 0; mt < 2; ++mt) {
            int m = wrow + mt * 16 + lm;
            afr[mt].q[0] = *(const u32x4*)&As[m][hi * 8];        // K 0..7 | 8..15
            afr[mt].q[1] = *(const u32x4*)&As[m][16 + hi * 8];   // K 16..23 | 24..31
        }
        #pragma unroll
        for (int nt = 0; nt < 4; ++nt) {
            int n = wcol + nt * 16 + lm;
            bfr[nt].q[0] = *(const u32x4*)&Bs[n][hi * 8];        // K 0..7 | 8..15
            bfr[nt].q[1] = *(const u32x4*)&Bs[n][16 + hi * 8];   // K 16..23 | 24..31
        }
        #pragma unroll
        for (int mt = 0; mt < 2; ++mt)
            #pragma unroll
            for (int nt = 0; nt < 4; ++nt)
                acc[mt][nt] = __builtin_amdgcn_wmma_f32_16x16x32_bf16(
                    false, afr[mt].v, false, bfr[nt].v,
                    (short)0, acc[mt][nt], false, false);

        __syncthreads();
    }

    // ---- epilogue: C/D layout = VGPR r -> M = hi*8 + r, N = lm
    #pragma unroll
    for (int mt = 0; mt < 2; ++mt) {
        #pragma unroll
        for (int nt = 0; nt < 4; ++nt) {
            int col = col0 + wcol + nt * 16 + lm;
            if (col >= N) continue;
            float bv = bias ? bias[col] : 0.0f;
            #pragma unroll
            for (int r = 0; r < 8; ++r) {
                int row = row0 + wrow + mt * 16 + hi * 8 + r;
                float v = acc[mt][nt][r] + bv;
                if (relu) v = fmaxf(v, 0.0f);
                if (Cf) Cf[(size_t)row * N + col] = v;
                if (Cb) Cb[(size_t)row * N + col] = f2bf(v);
            }
        }
    }
}

// ---------------------------------------------------------------------------
// concat [x | steps_left | current_r] -> bf16 [2048, 512]
// ---------------------------------------------------------------------------
__global__ void k_concat(const float* __restrict__ x, const float* __restrict__ sl,
                         const float* __restrict__ cr, unsigned short* __restrict__ o)
{
    int idx = blockIdx.x * 256 + threadIdx.x;
    if (idx >= MROWS * 512) return;
    int c = idx & 511, r = idx >> 9;
    float v;
    if (c < 510)       v = x[(size_t)r * 510 + c];
    else if (c == 510) v = sl[r];
    else               v = cr[r];
    o[idx] = f2bf(v);
}

// ---------------------------------------------------------------------------
// causal depthwise conv1d (k=4) + SiLU over the xBC slice of zxbcdt (fp32)
// ---------------------------------------------------------------------------
__global__ void k_conv_silu(const float* __restrict__ zx, const float* __restrict__ cw,
                            const float* __restrict__ cb, float* __restrict__ out)
{
    int idx = blockIdx.x * 256 + threadIdx.x;
    if (idx >= MROWS * CONVDIM) return;
    int c = idx % CONVDIM;
    int row = idx / CONVDIM;
    int l = row & (SEQLEN - 1);
    float acc = cb[c];
    #pragma unroll
    for (int k = 0; k < 4; ++k) {
        int dl = l + k - 3;
        float xv = (dl >= 0) ? zx[(size_t)(row + k - 3) * DINPROJ + DINNER + c] : 0.0f;
        acc += xv * cw[c * 4 + k];
    }
    out[(size_t)row * CONVDIM + c] = siluf(acc);
}

// ---------------------------------------------------------------------------
// dt = softplus(dt_raw + bias); dA = exp(dt * (-exp(A_log)))
// ---------------------------------------------------------------------------
__global__ void k_dt(const float* __restrict__ zx, const float* __restrict__ dtb,
                     const float* __restrict__ Alog,
                     float* __restrict__ dtS, float* __restrict__ dAv)
{
    int idx = blockIdx.x * 256 + threadIdx.x;
    if (idx >= MROWS * NHEADS) return;
    int h = idx & 31, row = idx >> 5;
    float xv = zx[(size_t)row * DINPROJ + (DINNER + CONVDIM) + h] + dtb[h];
    float dt = (xv > 20.0f) ? xv : log1pf(__expf(xv));
    float A  = -__expf(Alog[h]);
    dtS[idx] = dt;
    dAv[idx] = __expf(dt * A);
}

// ---------------------------------------------------------------------------
// selective scan: one WG per (b,h); state 64x128 f32 in registers
// (32 regs/thread). B/C rows (256 contiguous floats of xBC) are staged by
// double-buffered GLOBAL_LOAD_ASYNC_TO_LDS_B32: step l+1's copies are issued
// right after step l's barrier (so nobody still reads that buffer) and
// retired by s_wait_asynccnt 0 at the top of step l+1 -> staging latency
// hides behind the recurrence. thread t: p = t>>2, owns n in [(t&3)*32,+32);
// y-dot reduced via wave32 shfl_xor.
// ---------------------------------------------------------------------------
__global__ __launch_bounds__(256)
void k_scan(const float* __restrict__ xBC, const float* __restrict__ dtS,
            const float* __restrict__ dAv, const float* __restrict__ Dsk,
            float* __restrict__ y)
{
    __shared__ float BC[2][2 * DSTATE];   // [buf][ B(128) | C(128) ]
    const int bh = blockIdx.x;            // 0..127
    const int b  = bh >> 5;
    const int h  = bh & 31;
    const int t  = threadIdx.x;
    const int p  = t >> 2;
    const int ns = t & 3;
    const int nb = ns * 32;
    const float Dv = Dsk[h];

    const unsigned lds0 = (unsigned)(unsigned long long)(void*)&BC[0][t];
    const unsigned lds1 = (unsigned)(unsigned long long)(void*)&BC[1][t];
    const float* base = xBC + (size_t)(b * SEQLEN) * CONVDIM + DINNER; // B|C slice

    float s[32];
    #pragma unroll
    for (int j = 0; j < 32; ++j) s[j] = 0.0f;

    // prologue: stage step 0 into buffer 0
    async_ld_b32(lds0, base + t);

    for (int l = 0; l < SEQLEN; ++l) {
        const int row = b * SEQLEN + l;
        wait_async0();                    // current buffer's copies complete
        __syncthreads();                  // ... and visible to all waves
        if (l + 1 < SEQLEN)               // stage next step into other buffer
            async_ld_b32((l & 1) ? lds0 : lds1,
                         base + (size_t)(l + 1) * CONVDIM + t);

        const float* bc = BC[l & 1];
        float dt = dtS[row * NHEADS + h];
        float dA = dAv[row * NHEADS + h];
        float xp = xBC[(size_t)row * CONVDIM + h * HEADDIM + p];
        if (l + 1 < SEQLEN)
            __builtin_prefetch(&xBC[(size_t)(row + 1) * CONVDIM + h * HEADDIM + p], 0, 1);

        float coef = dt * xp;
        float part = 0.0f;
        #pragma unroll
        for (int j = 0; j < 32; ++j) {
            s[j] = s[j] * dA + coef * bc[nb + j];
            part += s[j] * bc[DSTATE + nb + j];
        }
        part += __shfl_xor(part, 1, 32);
        part += __shfl_xor(part, 2, 32);
        if (ns == 0)
            y[(size_t)row * DINNER + h * HEADDIM + p] = part + Dv * xp;
    }
}

// ---------------------------------------------------------------------------
// gated RMSNorm:  yb16 = ((y * silu(z)) * rsqrt(mean(sq)+eps)) * nw
// ---------------------------------------------------------------------------
__global__ __launch_bounds__(256)
void k_norm(const float* __restrict__ y, const float* __restrict__ zx,
            const float* __restrict__ nw, unsigned short* __restrict__ yb)
{
    __shared__ float red[8];
    const int row = blockIdx.x;
    const int t = threadIdx.x;
    float v[8];
    float ss = 0.0f;
    #pragma unroll
    for (int i = 0; i < 8; ++i) {
        int c = t + i * 256;
        float g   = siluf(zx[(size_t)row * DINPROJ + c]);   // z slice
        float val = y[(size_t)row * DINNER + c] * g;
        v[i] = val;
        ss += val * val;
    }
    #pragma unroll
    for (int m = 16; m >= 1; m >>= 1) ss += __shfl_xor(ss, m, 32);
    if ((t & 31) == 0) red[t >> 5] = ss;
    __syncthreads();
    float tot = 0.0f;
    #pragma unroll
    for (int i = 0; i < 8; ++i) tot += red[i];
    float scale = rsqrtf(tot * (1.0f / DINNER) + 1e-5f);
    #pragma unroll
    for (int i = 0; i < 8; ++i) {
        int c = t + i * 256;
        yb[(size_t)row * DINNER + c] = f2bf(v[i] * scale * nw[c]);
    }
}

// ---------------------------------------------------------------------------
// final 1024 -> 1 projection (bf16 activations, fp32 weights/accum)
// ---------------------------------------------------------------------------
__global__ __launch_bounds__(256)
void k_outdot(const unsigned short* __restrict__ a, const float* __restrict__ w,
              const float* __restrict__ b, float* __restrict__ out)
{
    __shared__ float red[8];
    const int row = blockIdx.x;
    const int t = threadIdx.x;
    float ss = 0.0f;
    #pragma unroll
    for (int i = 0; i < 4; ++i) {
        int c = t + i * 256;
        ss += bf2f(a[(size_t)row * DMODEL + c]) * w[c];
    }
    #pragma unroll
    for (int m = 16; m >= 1; m >>= 1) ss += __shfl_xor(ss, m, 32);
    if ((t & 31) == 0) red[t >> 5] = ss;
    __syncthreads();
    if (t == 0) {
        float tot = 0.0f;
        #pragma unroll
        for (int i = 0; i < 8; ++i) tot += red[i];
        out[row] = tot + b[0];
    }
}

// ---------------------------------------------------------------------------
extern "C" void kernel_launch(void* const* d_in, const int* in_sizes, int n_in,
                              void* d_out, int out_size, void* d_ws, size_t ws_size,
                              hipStream_t stream)
{
    const float* x    = (const float*)d_in[0];
    const float* sl   = (const float*)d_in[1];
    const float* cr   = (const float*)d_in[2];
    const float* iw1  = (const float*)d_in[3];
    const float* ib1  = (const float*)d_in[4];
    const float* iw2  = (const float*)d_in[5];
    const float* ib2  = (const float*)d_in[6];
    const float* Wi   = (const float*)d_in[7];   // [4,1024,4384]
    const float* cw   = (const float*)d_in[8];   // [4,2304,4]
    const float* cb   = (const float*)d_in[9];   // [4,2304]
    const float* dtb  = (const float*)d_in[10];  // [4,32]
    const float* Alog = (const float*)d_in[11];  // [4,32]
    const float* Dsk  = (const float*)d_in[12];  // [4,32]
    const float* nw   = (const float*)d_in[13];  // [4,2048]
    const float* Wo   = (const float*)d_in[14];  // [4,2048,1024]
    const float* ow1  = (const float*)d_in[15];
    const float* ob1  = (const float*)d_in[16];
    const float* ow2  = (const float*)d_in[17];
    const float* ob2  = (const float*)d_in[18];
    float* out = (float*)d_out;

    // workspace carve-out (~99 MB), 256-B aligned
    char* ws = (char*)d_ws;
    size_t off = 0;
    auto carve = [&](size_t bytes) -> void* {
        void* p = ws + off;
        off += (bytes + 255) & ~(size_t)255;
        return p;
    };
    typedef unsigned short u16;
    u16*  T0b = (u16*) carve((size_t)MROWS * 512     * 2);  // concat (bf16)
    u16*  Hb  = (u16*) carve((size_t)MROWS * DMODEL  * 2);  // hidden (bf16)
    u16*  M1b = (u16*) carve((size_t)MROWS * DMODEL  * 2);  // MLP temp (bf16)
    u16*  Yb  = (u16*) carve((size_t)MROWS * DINNER  * 2);  // normed y (bf16)
    u16*  WB  = (u16*) carve((size_t)DMODEL * DINPROJ * 2 + 4096); // padded bf16 W scratch
    float* ZX  = (float*)carve((size_t)MROWS * DINPROJ * 4); // zxbcdt (fp32)
    float* XBC = (float*)carve((size_t)MROWS * CONVDIM * 4);
    float* DTS = (float*)carve((size_t)MROWS * NHEADS  * 4);
    float* DAV = (float*)carve((size_t)MROWS * NHEADS  * 4);
    float* Y   = (float*)carve((size_t)MROWS * DINNER  * 4);
    (void)ws_size; (void)in_sizes; (void)n_in; (void)out_size;

    const dim3 blk(256);
    const int MT = MROWS / TM;   // 16 row tiles
    auto cvt = [&](const float* w, int nelem) {
        int n8 = nelem / 8;
        k_f2bf<<<(n8 + 255) / 256, blk, 0, stream>>>(w, WB, n8);
    };

    // input MLP: Linear(512,512)+ReLU -> Linear(512,1024)
    k_concat<<<(MROWS * 512 + 255) / 256, blk, 0, stream>>>(x, sl, cr, T0b);
    cvt(iw1, 512 * 512);
    k_gemm_bf16<<<dim3(512 / TN, MT), blk, 0, stream>>>(
        T0b, WB, ib1, nullptr, M1b, MROWS, 512, 512, 1);
    cvt(iw2, 512 * DMODEL);
    k_gemm_bf16<<<dim3(DMODEL / TN, MT), blk, 0, stream>>>(
        M1b, WB, ib2, nullptr, Hb, MROWS, DMODEL, 512, 0);

    for (int i = 0; i < 4; ++i) {
        cvt(Wi + (size_t)i * DMODEL * DINPROJ, DMODEL * DINPROJ);
        k_gemm_bf16<<<dim3((DINPROJ + TN - 1) / TN, MT), blk, 0, stream>>>(
            Hb, WB, nullptr, ZX, nullptr, MROWS, DINPROJ, DMODEL, 0);
        k_conv_silu<<<(MROWS * CONVDIM + 255) / 256, blk, 0, stream>>>(
            ZX, cw + (size_t)i * CONVDIM * 4, cb + (size_t)i * CONVDIM, XBC);
        k_dt<<<(MROWS * NHEADS + 255) / 256, blk, 0, stream>>>(
            ZX, dtb + i * NHEADS, Alog + i * NHEADS, DTS, DAV);
        k_scan<<<4 * NHEADS, blk, 0, stream>>>(XBC, DTS, DAV, Dsk + i * NHEADS, Y);
        k_norm<<<MROWS, blk, 0, stream>>>(Y, ZX, nw + (size_t)i * DINNER, Yb);
        cvt(Wo + (size_t)i * DINNER * DMODEL, DINNER * DMODEL);
        k_gemm_bf16<<<dim3(DMODEL / TN, MT), blk, 0, stream>>>(
            Yb, WB, nullptr, nullptr, Hb, MROWS, DMODEL, DINNER, 0);
    }

    // output MLP: Linear(1024,1024)+ReLU -> Linear(1024,1)
    cvt(ow1, DMODEL * DMODEL);
    k_gemm_bf16<<<dim3(DMODEL / TN, MT), blk, 0, stream>>>(
        Hb, WB, ob1, nullptr, M1b, MROWS, DMODEL, DMODEL, 1);
    k_outdot<<<MROWS, blk, 0, stream>>>(M1b, ow2, ob2, out);
}
